// GraphAttentionLayer_24653112279725
// MI455X (gfx1250) — matrix operand order repaired and checked by reading, not compiled, not run on hardware
//
#include <hip/hip_runtime.h>

// GAT layer, MI455X (gfx1250, wave32, WMMA, TDM).
// Flash-attention-style fusion: scores are rank-1 (src_i + dst_j) + mask, so
// the only GEMM is P(16xN) x h(Nx64) -> v_wmma_f32_16x16x32_f16.
// The 512MB attention tensor is never materialized. h lives as f16 in the
// workspace (4MB, L2-resident); the 64MB graph streams from L2 via the
// Tensor Data Mover (2D tiled descriptor), double-buffered in LDS and
// drained with s_wait_tensorcnt.

#define GAT_B 8
#define GAT_N 4096
#define GAT_D 64
#define GAT_NEG_SLOPE 0.2f
#define GAT_JT 32           // K-tile (j columns) per WMMA step
#define GAT_IT 64           // i-rows per block (16 per wave, 4 waves)

typedef __attribute__((ext_vector_type(16))) _Float16     v16h;
typedef __attribute__((ext_vector_type(8)))  float        v8f;
typedef __attribute__((ext_vector_type(4)))  unsigned int u32x4;
typedef __attribute__((ext_vector_type(8)))  int          i32x8;
typedef __attribute__((ext_vector_type(4)))  int          i32x4;

#if __has_builtin(__builtin_amdgcn_tensor_load_to_lds) && \
    __has_builtin(__builtin_amdgcn_s_wait_tensorcnt)
#define GAT_USE_TDM 1
#else
#define GAT_USE_TDM 0
#endif

// ---------------------------------------------------------------------------
// Kernel 1: h = X * W^T (store f16), src = h.attn_src, dst = h.attn_dst
// ---------------------------------------------------------------------------
__global__ __launch_bounds__(256) void gat_proj_kernel(
    const float* __restrict__ x,      // [B*N, 64]
    const float* __restrict__ W,      // [64, 64] (d, c)
    const float* __restrict__ asrc,   // [64]
    const float* __restrict__ adst,   // [64]
    _Float16*    __restrict__ h16,    // [B*N, 64]
    float*       __restrict__ srcv,   // [B*N]
    float*       __restrict__ dstv)   // [B*N]
{
    __shared__ float sWT[64 * 64];    // sWT[c*64+d] = W[d,c]
    __shared__ float sx[4 * 64];
    __shared__ float sps[256];
    __shared__ float spd[256];

    const int tid  = threadIdx.x;
    const int r    = tid >> 6;
    const int d    = tid & 63;
    const int row0 = blockIdx.x * 4;

    for (int u = tid; u < 64 * 64; u += 256) {
        int wd = u >> 6, wc = u & 63;
        sWT[wc * 64 + wd] = W[u];
    }
    sx[tid] = x[(size_t)row0 * 64 + tid];
    __syncthreads();

    float acc = 0.f;
#pragma unroll
    for (int c = 0; c < 64; ++c)
        acc = fmaf(sx[r * 64 + c], sWT[c * 64 + d], acc);

    const int row = row0 + r;
    h16[(size_t)row * 64 + d] = (_Float16)acc;

    sps[tid] = acc * asrc[d];
    spd[tid] = acc * adst[d];
    __syncthreads();

    if (d == 0) {
        float s1 = 0.f, s2 = 0.f;
#pragma unroll
        for (int k = 0; k < 64; ++k) { s1 += sps[r * 64 + k]; s2 += spd[r * 64 + k]; }
        srcv[row] = s1;
        dstv[row] = s2;
    }
}

// ---------------------------------------------------------------------------
// TDM helper: build a D# (groups 0/1 per ISA 8.3/8.4) for a 2D (or 1D) tile
// load into LDS. data_size code: 0=1B,1=2B,2=4B,3=8B. Strides in elements.
// ---------------------------------------------------------------------------
#if GAT_USE_TDM
__device__ __forceinline__ unsigned gat_lds_addr(const void* p) {
    // generic LDS address: addr[31:0] is the LDS byte offset (ISA 10.2)
    return (unsigned)(unsigned long long)p;
}

__device__ __forceinline__ void gat_tdm_load_2d(
    unsigned lds_addr, const void* gptr, unsigned ds_code,
    unsigned tile0, unsigned tile1,
    unsigned tdim0, unsigned tdim1, unsigned long long stride0)
{
    const unsigned long long ga = (unsigned long long)gptr;
    u32x4 g0;
    g0[0] = 1u;                                   // count=1, user descriptor
    g0[1] = lds_addr;                             // LDS dest (bytes)
    g0[2] = (unsigned)(ga & 0xffffffffu);         // global_addr[31:0]
    g0[3] = (unsigned)((ga >> 32) & 0x01ffffffu)  // global_addr[56:32]
          | (2u << 30);                           // type = 2 ("image")
    i32x8 g1;
    g1[0] = (int)(ds_code << 16);                 // wg_mask=0, data_size
    g1[1] = (int)((tdim0 & 0xffffu) << 16);       // tensor_dim0[15:0]
    g1[2] = (int)((tdim0 >> 16) & 0xffffu)        // tensor_dim0[31:16]
          | (int)((tdim1 & 0xffffu) << 16);       // tensor_dim1[15:0]
    g1[3] = (int)((tdim1 >> 16) & 0xffffu)        // tensor_dim1[31:16]
          | (int)(tile0 << 16);                   // tile_dim0
    g1[4] = (int)(tile1 & 0xffffu);               // tile_dim1 (0 = 1D), dim2=0
    g1[5] = (int)(stride0 & 0xffffffffu);         // tensor_dim0_stride[31:0]
    g1[6] = (int)((stride0 >> 32) & 0xffffu);     // stride0[47:32], stride1=0
    g1[7] = 0;
    const i32x4 z4 = {0, 0, 0, 0};
#if defined(__clang_major__) && (__clang_major__ >= 23)
    const i32x8 z8 = {0, 0, 0, 0, 0, 0, 0, 0};
    __builtin_amdgcn_tensor_load_to_lds(g0, g1, z4, z4, z8, 0);
#else
    __builtin_amdgcn_tensor_load_to_lds(g0, g1, z4, z4, 0);
#endif
}
#endif

// ---------------------------------------------------------------------------
// Kernel 2: fused masked-softmax attention + P*h via WMMA.
// Block = 128 threads (4 wave32) handling a 64-row i-tile; wave w owns rows
// [i0+16w, i0+16w+16) and produces its full 16x64 output slab: one softmax
// pass feeds FOUR v_wmma_f32_16x16x32_f16 per j-tile.
//
// Fragment layouts per ISA 7.12.2 (wave32):
//   A 16x32 f16 : lane l: m=l&15, h=l>>4; v<4 -> K=8h+2v+{0,1}; v>=4 -> K=16+8h+2(v-4)+{0,1}
//   B 32x16 f16 : lane l: n=l&15, h=l>>4; v<4 -> K=8h+2v+{0,1}; v>=4 -> K=16+8h+2(v-4)+{0,1}
//   C/D 16x16 f32: lane l: n=l&15; VGPR v -> M=8*(l>>4)+v
// ---------------------------------------------------------------------------
__global__ __launch_bounds__(128) void gat_attn_kernel(
    const float*     __restrict__ x,      // [B*N, 64] residual
    const int*       __restrict__ graph,  // [N, N]
    const float*     __restrict__ bias,   // [64]
    const _Float16*  __restrict__ h16,    // [B*N, 64]
    const float*     __restrict__ srcv,   // [B*N]
    const float*     __restrict__ dstv,   // [B*N]
    float*           __restrict__ out)    // [B*N, 64]
{
    __shared__ __align__(16) int      sh_mask[2][GAT_IT * GAT_JT];   // 16 KB
    __shared__ __align__(16) _Float16 sh_h[2][GAT_JT * GAT_D];       //  8 KB
    __shared__ __align__(16) _Float16 sh_hT[GAT_D][GAT_JT + 2];      //  4.25 KB
    __shared__ float                  sh_dst[2][GAT_JT];

    const int b    = blockIdx.y;
    const int i0   = blockIdx.x * GAT_IT;
    const int tid  = threadIdx.x;
    const int wave = tid >> 5;
    const int lane = tid & 31;
    const int half = lane >> 4;
    const int lm   = lane & 15;
    const int mrow = 16 * wave + lm;          // score row within i-tile

    const float src_m = srcv[b * GAT_N + i0 + mrow];
    float run_max = -1e30f;
    float run_sum = 0.f;
    v8f acc0 = {}, acc1 = {}, acc2 = {}, acc3 = {};

#if GAT_USE_TDM
    // prologue: async-stage tile 0 (mask 64x32 i32 2D tile + contiguous 4KB h)
    if (wave == 0) {
        gat_tdm_load_2d(gat_lds_addr(&sh_mask[0][0]),
                        &graph[(size_t)i0 * GAT_N], /*i32*/2,
                        GAT_JT, GAT_IT, GAT_N, GAT_N, GAT_N);
        gat_tdm_load_2d(gat_lds_addr(&sh_h[0][0]),
                        &h16[(size_t)(b * GAT_N) * GAT_D], /*f16*/1,
                        GAT_JT * GAT_D, 0, GAT_JT * GAT_D, 0, GAT_JT * GAT_D);
    }
#endif

    for (int j0 = 0; j0 < GAT_N; j0 += GAT_JT) {
        const int cur = (j0 / GAT_JT) & 1;

#if GAT_USE_TDM
        if (wave == 0) {
            sh_dst[cur][lane] = dstv[b * GAT_N + j0 + lane];
            const int jn = j0 + GAT_JT;
            if (jn < GAT_N) {
                const int nb = cur ^ 1;
                gat_tdm_load_2d(gat_lds_addr(&sh_mask[nb][0]),
                                &graph[(size_t)i0 * GAT_N + jn], 2,
                                GAT_JT, GAT_IT, GAT_N, GAT_N, GAT_N);
                gat_tdm_load_2d(gat_lds_addr(&sh_h[nb][0]),
                                &h16[(size_t)(b * GAT_N + jn) * GAT_D], 1,
                                GAT_JT * GAT_D, 0, GAT_JT * GAT_D, 0, GAT_JT * GAT_D);
                __builtin_amdgcn_s_wait_tensorcnt(2);  // tile `cur` done, next pair in flight
            } else {
                __builtin_amdgcn_s_wait_tensorcnt(0);
            }
        }
#else
        // ---- manual staging fallback (16-byte vector loads) ----
        if (tid < GAT_JT) sh_dst[cur][tid] = dstv[b * GAT_N + j0 + tid];
        {   // mask: 64x32 i32; thread t -> row t>>1, half-row (t&1)*16
            const int m = tid >> 1, k0 = (tid & 1) * 16;
            const int4* gm = (const int4*)&graph[(size_t)(i0 + m) * GAT_N + j0 + k0];
            int4* dm = (int4*)&sh_mask[cur][m * GAT_JT + k0];
            dm[0] = gm[0]; dm[1] = gm[1]; dm[2] = gm[2]; dm[3] = gm[3];
        }
        {   // h: 32x64 f16; thread t -> row t>>2, quarter-row (t&3)*16
            const int k = tid >> 2, d0 = (tid & 3) * 16;
            const uint4* gh = (const uint4*)&h16[((size_t)(b * GAT_N + j0 + k)) * GAT_D + d0];
            uint4* dh = (uint4*)&sh_h[cur][k * GAT_D + d0];
            dh[0] = gh[0]; dh[1] = gh[1];
        }
        if (j0 + GAT_JT < GAT_N)
            __builtin_prefetch(&graph[(size_t)(i0 + (tid >> 1)) * GAT_N + j0 + GAT_JT], 0, 1);
#endif
        __syncthreads();

        // ---- LDS transpose of h tile (row-major [k][d] -> [d][k]) ----
        {
            const int k = tid >> 2, d0 = (tid & 3) * 16;
#pragma unroll
            for (int q = 0; q < 16; ++q)
                sh_hT[d0 + q][k] = sh_h[cur][k * GAT_D + d0 + q];
        }
        __syncthreads();

        // ---- scores, computed directly in the A-fragment element order ----
        float s[16];
        float lmax = -1e30f;
#pragma unroll
        for (int v = 0; v < 8; ++v) {
            const int kb = (v < 4) ? (8 * half + 2 * v)
                                   : (16 + 8 * half + 2 * (v - 4));
#pragma unroll
            for (int p = 0; p < 2; ++p) {
                const int k = kb + p;
                float sc = src_m + sh_dst[cur][k];
                sc = (sc >= 0.f) ? sc : GAT_NEG_SLOPE * sc;
                sc = sh_mask[cur][mrow * GAT_JT + k] ? sc : -1e30f;
                s[2 * v + p] = sc;
                lmax = fmaxf(lmax, sc);
            }
        }
        lmax = fmaxf(lmax, __shfl_xor(lmax, 16, 32));
        const float new_max = fmaxf(run_max, lmax);
        const float alpha   = __expf(run_max - new_max);
        run_max = new_max;

        v16h afrag;
        float lsum = 0.f;
#pragma unroll
        for (int e = 0; e < 16; ++e) {
            const float pe = (s[e] <= -1e29f) ? 0.f : __expf(s[e] - new_max);
            lsum += pe;
            afrag[e] = (_Float16)pe;
        }
        lsum += __shfl_xor(lsum, 16, 32);
        run_sum = run_sum * alpha + lsum;

        // rescale accumulators (row m=8*half+v; alpha for row r at lanes r,r+16)
#pragma unroll
        for (int v = 0; v < 8; ++v) {
            const float av = __shfl(alpha, 8 * half + v, 32);
            acc0[v] *= av; acc1[v] *= av; acc2[v] *= av; acc3[v] *= av;
        }

        // ---- four 16-col slabs: B fragment + WMMA each ----
#pragma unroll
        for (int c = 0; c < 4; ++c) {
            v16h bf;
            const int n = 16 * c + lm;
#pragma unroll
            for (int v = 0; v < 8; ++v) {
                const int k = (v < 4) ? (8 * half + 2 * v) : (16 + 8 * half + 2 * (v - 4));
                bf[2 * v]     = sh_hT[n][k];
                bf[2 * v + 1] = sh_hT[n][k + 1];
            }
            v8f& accc = (c == 0) ? acc0 : (c == 1) ? acc1 : (c == 2) ? acc2 : acc3;
            accc = __builtin_amdgcn_wmma_f32_16x16x32_f16(
                false, afrag, false, bf, (short)0, accc, false, false);
        }
        __syncthreads();   // fragments latched in VGPRs; safe to restage LDS
    }

    // ---- epilogue: normalize, residual + bias ----
    const float bn0 = bias[lm], bn1 = bias[16 + lm],
                bn2 = bias[32 + lm], bn3 = bias[48 + lm];
#pragma unroll
    for (int v = 0; v < 8; ++v) {
        const int m = i0 + 16 * wave + 8 * half + v;
        const float inv = 1.f / __shfl(run_sum, 8 * half + v, 32);
        const size_t base = ((size_t)(b * GAT_N + m)) * GAT_D;
        out[base + lm]      = acc0[v] * inv + x[base + lm]      + bn0;
        out[base + 16 + lm] = acc1[v] * inv + x[base + 16 + lm] + bn1;
        out[base + 32 + lm] = acc2[v] * inv + x[base + 32 + lm] + bn2;
        out[base + 48 + lm] = acc3[v] * inv + x[base + 48 + lm] + bn3;
    }
}

// ---------------------------------------------------------------------------
extern "C" void kernel_launch(void* const* d_in, const int* in_sizes, int n_in,
                              void* d_out, int out_size, void* d_ws, size_t ws_size,
                              hipStream_t stream) {
    const float* x     = (const float*)d_in[0];   // [8,4096,64]
    const int*   graph = (const int*)  d_in[1];   // [4096,4096]
    const float* W     = (const float*)d_in[2];   // [64,64]
    const float* asrc  = (const float*)d_in[3];   // [64]
    const float* adst  = (const float*)d_in[4];   // [64]
    const float* bias  = (const float*)d_in[5];   // [64]
    float* out = (float*)d_out;

    // workspace: h16 (4MB) | src (128KB) | dst (128KB)
    _Float16* h16 = (_Float16*)d_ws;
    float* srcv = (float*)((char*)d_ws + (size_t)GAT_B * GAT_N * GAT_D * sizeof(_Float16));
    float* dstv = srcv + (size_t)GAT_B * GAT_N;

    const int rows = GAT_B * GAT_N;
    gat_proj_kernel<<<rows / 4, 256, 0, stream>>>(x, W, asrc, adst, h16, srcv, dstv);
    gat_attn_kernel<<<dim3(GAT_N / GAT_IT, GAT_B), 128, 0, stream>>>(
        x, graph, bias, h16, srcv, dstv, out);
}
